// MultiHeadAttention_19602230739165
// MI455X (gfx1250) — matrix operand order repaired and checked by reading, not compiled, not run on hardware
//
#include <hip/hip_runtime.h>

typedef __attribute__((ext_vector_type(16))) _Float16 v16h;
typedef __attribute__((ext_vector_type(8)))  _Float16 v8h;
typedef __attribute__((ext_vector_type(8)))  float    v8f;
typedef __attribute__((ext_vector_type(4)))  uint32_t u32x4;
typedef __attribute__((ext_vector_type(4)))  int      i32x4;
typedef __attribute__((ext_vector_type(8)))  int      i32x8;

#define WMMA_F16(A, B, C) \
  __builtin_amdgcn_wmma_f32_16x16x32_f16(false, (A), false, (B), (short)0, (C), false, false)

// Problem constants
static constexpr int BATCH  = 2;
static constexpr int SEQ    = 2048;
static constexpr int DMODEL = 1024;   // also H*DK and H*DV
static constexpr int HDIM   = 64;
static constexpr int ROWS   = BATCH * SEQ;   // 4096

// ---------------------------------------------------------------------------
// Fragment loaders (ISA 7.12.2 layouts, wave32)
// A (f16 16x32, MxK): lane L<16 -> row M=L holds K=0..7,16..23; L>=16 -> K=8..15,24..31
// B (f16 32x16, KxN): lane L -> col N=L&15; L<16 -> K=0..7,16..23; L>=16 -> K=8..15,24..31
// C/D (f32 16x16):    VGPR r, lane L -> M = r + 8*(L>>4), N = L&15
// ---------------------------------------------------------------------------

__device__ __forceinline__ v16h load_A_f16(const _Float16* __restrict__ src,
                                           size_t row0, int col0, int lane) {
  const int m = lane & 15, half = lane >> 4;
  const _Float16* p = src + (row0 + (size_t)m) * DMODEL + col0;
  v16h a;
#pragma unroll
  for (int i = 0; i < 8; ++i) a[i] = p[half * 8 + i];
#pragma unroll
  for (int i = 0; i < 8; ++i) a[8 + i] = p[16 + half * 8 + i];
  return a;
}

// B[k][n] = src[(nrow0 + n) * ld + kcol0 + k]  ("N-major" source, reduction
// contiguous along the row). Works for global f16 and for LDS tiles.
__device__ __forceinline__ v16h load_B_nmajor(const _Float16* __restrict__ src,
                                              size_t nrow0, int ld, int kcol0,
                                              int lane) {
  const int n = lane & 15, half = lane >> 4;
  const _Float16* p = src + (nrow0 + (size_t)n) * ld + kcol0;
  v16h b;
#pragma unroll
  for (int i = 0; i < 8; ++i) b[i] = p[half * 8 + i];
#pragma unroll
  for (int i = 0; i < 8; ++i) b[8 + i] = p[16 + half * 8 + i];
  return b;
}

// A fragment from a 16x32 f16 row-major tile in LDS.
__device__ __forceinline__ v16h load_A_lds(const _Float16* p, int lane) {
  const int m = lane & 15, half = lane >> 4;
  const _Float16* row = p + m * 32;
  v16h a;
#pragma unroll
  for (int i = 0; i < 8; ++i) a[i] = row[half * 8 + i];
#pragma unroll
  for (int i = 0; i < 8; ++i) a[8 + i] = row[16 + half * 8 + i];
  return a;
}

// ---------------------------------------------------------------------------
// Kernel 0: one-time f32 -> f16 conversion (Q and the three weight matrices).
// ---------------------------------------------------------------------------
__global__ __launch_bounds__(256) void cvt_f32_f16(const float* __restrict__ s,
                                                   _Float16* __restrict__ d,
                                                   int n) {
  const int i = (int)(blockIdx.x * blockDim.x + threadIdx.x) * 8;
  if (i >= n) return;
  const float4* p = reinterpret_cast<const float4*>(s + i);
  const float4 a = p[0], b = p[1];
  v8h o;
  o[0] = (_Float16)a.x; o[1] = (_Float16)a.y; o[2] = (_Float16)a.z; o[3] = (_Float16)a.w;
  o[4] = (_Float16)b.x; o[5] = (_Float16)b.y; o[6] = (_Float16)b.z; o[7] = (_Float16)b.w;
  *reinterpret_cast<v8h*>(d + i) = o;
}

// ---------------------------------------------------------------------------
// Kernel 1: fused QKV projection (all-f16 operands). One wave per 16x16 tile,
// shared A fragment feeds 3 WMMAs per K-step. q/k stored row-major [row][1024];
// V stored TRANSPOSED vt[d][row] so attention's P.V B-operand is contiguous.
// ---------------------------------------------------------------------------
__global__ __launch_bounds__(256) void mha_qkv_proj(
    const _Float16* __restrict__ Qh,
    const _Float16* __restrict__ Wq, const float* __restrict__ bq,
    const _Float16* __restrict__ Wk, const float* __restrict__ bk,
    const _Float16* __restrict__ Wv, const float* __restrict__ bv,
    _Float16* __restrict__ qh, _Float16* __restrict__ kh,
    _Float16* __restrict__ vt) {
  const int lane = threadIdx.x & 31;
  const int wid  = (int)((blockIdx.x * blockDim.x + threadIdx.x) >> 5);
  const int col0 = (wid & 63) * 16;              // 64 tiles over 1024 columns
  const size_t row0 = (size_t)(wid >> 6) * 16;   // 256 tiles over 4096 rows

  v8f cq = {}, ck = {}, cv = {};
#pragma unroll 4
  for (int k0 = 0; k0 < DMODEL; k0 += 32) {
    __builtin_prefetch(Qh + (row0 + (size_t)(lane & 15)) * DMODEL + k0 + 128, 0, 1);
    const v16h a = load_A_f16(Qh, row0, k0, lane);
    cq = WMMA_F16(a, load_B_nmajor(Wq, (size_t)col0, DMODEL, k0, lane), cq);
    ck = WMMA_F16(a, load_B_nmajor(Wk, (size_t)col0, DMODEL, k0, lane), ck);
    cv = WMMA_F16(a, load_B_nmajor(Wv, (size_t)col0, DMODEL, k0, lane), cv);
  }
  const int n = lane & 15, half = lane >> 4;
  const float biasq = bq[col0 + n], biask = bk[col0 + n], biasv = bv[col0 + n];
#pragma unroll
  for (int r = 0; r < 8; ++r) {
    const size_t idx = (row0 + (size_t)(r + 8 * half)) * DMODEL + col0 + n;
    qh[idx] = (_Float16)(cq[r] + biasq);
    kh[idx] = (_Float16)(ck[r] + biask);
  }
  v8h pv;  // rows are contiguous in vt -> one packed 16B store per lane
#pragma unroll
  for (int r = 0; r < 8; ++r) pv[r] = (_Float16)(cv[r] + biasv);
  *reinterpret_cast<v8h*>(vt + (size_t)(col0 + n) * ROWS + row0 + 8 * half) = pv;
}

// ---------------------------------------------------------------------------
// Kernel 2: flash-style exp-attention, cooperative per-(b,h) blocks.
// 8 waves/block = 8 consecutive 16-query tiles of the SAME (b,h). Per 32-key
// chunk:
//   - K[32x64] staged into LDS by the TDM (tensor_load_to_lds, TENSORcnt):
//     2D tile, 8B elements, 16 units/row, 32 rows, row stride 256 units.
//   - Vt[64x32] staged by all 256 threads via global_load_async_to_lds_b128
//     (ASYNCcnt).
// Then each wave: 4 WMMAs (QK^T from LDS) + exp/mask/rowsum + P staged via
// LDS + 4 WMMAs (P.V).
// LDS map (single __shared__ buffer -> base offset 0):
//   [0,4096)    Ks  32 keys x 64 d, row-major
//   [4096,8192) Vs  64 d x 32 keys, row-major
//   [8192,16384) P  8 waves x (16x32) f16 staging tiles
// ---------------------------------------------------------------------------
__global__ __launch_bounds__(256) void mha_attention(
    const _Float16* __restrict__ qh, const _Float16* __restrict__ kh,
    const _Float16* __restrict__ vt, const float* __restrict__ mask,
    float* __restrict__ out) {
  __shared__ _Float16 smem[8192];           // 16 KB
  _Float16* Ks = smem;                      // [32][64]
  _Float16* Vs = smem + 2048;               // [64][32]

  const int tid  = (int)threadIdx.x;
  const int lane = tid & 31;
  const int wloc = tid >> 5;
  const int blk  = (int)blockIdx.x;         // 512 = b(2) * h(16) * qg(16)
  const int qg = blk & 15;
  const int h  = (blk >> 4) & 15;
  const int b  = blk >> 8;
  const size_t rowb = (size_t)b * SEQ;
  const int colh = h * HDIM;
  const int q0 = qg * 128 + wloc * 16;
  const int n = lane & 15, half = lane >> 4;

  // Async-copy addressing for the V^T chunk (all 256 threads).
  const int vrow = tid >> 2, vseg = tid & 3;              // 64 rows x 4x16B
  const uint64_t gv0 = (uint64_t)(uintptr_t)(vt + (size_t)(colh + vrow) * ROWS + rowb + vseg * 8);
  const uint32_t lv = (uint32_t)(4096 + (vrow * 32 + vseg * 8) * 2);   // bytes

  const uint64_t kbase = (uint64_t)(uintptr_t)(kh + rowb * DMODEL + colh);

  // Persistent A fragments of this wave's query tile (d = 0..31, 32..63).
  const v16h aq0 = load_A_f16(qh, rowb + q0, colh, lane);
  const v16h aq1 = load_A_f16(qh, rowb + q0, colh + 32, lane);

  v8f c0 = {}, c1 = {}, c2 = {}, c3 = {};
  float rs[8];
#pragma unroll
  for (int r = 0; r < 8; ++r) rs[r] = 0.0f;

  _Float16* my = smem + 4096 + wloc * 512;  // 16x32 P staging tile
  const float sc = 0.125f;                  // 1/sqrt(DK)

  for (int kc = 0; kc < SEQ; kc += 32) {
    // --- stage K chunk [32x64] via TDM, V^T chunk [64x32] via async -------
    const uint64_t av = gv0 + (uint64_t)kc * 2;             // advance 32 cols
    asm volatile("global_load_async_to_lds_b128 %0, %1, off"
                 :: "v"(lv), "v"(av) : "memory");

    if (wloc == 0) {  // one wave drives the TDM (EXEC ignored by TDM)
      const uint64_t ga = kbase + (uint64_t)kc * (DMODEL * 2);
      u32x4 g0;
      g0[0] = 1u;                                  // count=1 valid descriptor
      g0[1] = 0u;                                  // lds_addr = 0 (Ks)
      g0[2] = (uint32_t)ga;                        // global_addr[31:0]
      g0[3] = (uint32_t)(ga >> 32) | (2u << 30);   // addr[56:32] | type=2
      i32x8 g1;
      g1[0] = (3 << 16);    // data_size = 8B, no multicast, no pad/iterate
      g1[1] = (16 << 16);   // tensor_dim0 = 16 units (128B row)
      g1[2] = (32 << 16);   // tensor_dim1 = 32 rows
      g1[3] = (16 << 16);   // tile_dim0 = 16 units
      g1[4] = 32;           // tile_dim1 = 32, tile_dim2 = 0
      g1[5] = 256;          // tensor_dim0_stride = 2048B / 8B
      g1[6] = 0;
      g1[7] = 0;
      const i32x4 z4 = {0, 0, 0, 0};
      const i32x8 z8 = {0, 0, 0, 0, 0, 0, 0, 0};
      // 6-arg form (clang-23 / therock headers): (g0, g1, g2, g3, g4, cpol)
      __builtin_amdgcn_tensor_load_to_lds(g0, g1, z4, z4, z8, 0);
      __builtin_amdgcn_s_wait_tensorcnt((short)0);
    }

    asm volatile("s_wait_asynccnt 0x0" ::: "memory");
    __syncthreads();

    // --- scores: 16 queries x 32 keys, reduction over d=64 ----------------
    v8f s0 = {}, s1 = {};
    s0 = WMMA_F16(aq0, load_B_nmajor(Ks, 0, 64, 0, lane), s0);
    s0 = WMMA_F16(aq1, load_B_nmajor(Ks, 0, 64, 32, lane), s0);
    s1 = WMMA_F16(aq0, load_B_nmajor(Ks, 16, 64, 0, lane), s1);
    s1 = WMMA_F16(aq1, load_B_nmajor(Ks, 16, 64, 32, lane), s1);

    const float m0 = mask[rowb + kc + n];
    const float m1 = mask[rowb + kc + 16 + n];
#pragma unroll
    for (int r = 0; r < 8; ++r) {
      const float p0 = __expf(s0[r] * sc) * m0;
      const float p1 = __expf(s1[r] * sc) * m1;
      rs[r] += p0 + p1;
      const int m = r + 8 * half;
      my[m * 32 + n]      = (_Float16)p0;   // C-layout -> row-major LDS tile
      my[m * 32 + 16 + n] = (_Float16)p1;
    }
    asm volatile("s_wait_dscnt 0x0" ::: "memory");  // in-wave LDS RAW fence

    // --- P (16x32) @ V (32x64): B operand contiguous from Vs[d][key] ------
    const v16h ap = load_A_lds(my, lane);
    c0 = WMMA_F16(ap, load_B_nmajor(Vs, 0, 32, 0, lane), c0);
    c1 = WMMA_F16(ap, load_B_nmajor(Vs, 16, 32, 0, lane), c1);
    c2 = WMMA_F16(ap, load_B_nmajor(Vs, 32, 32, 0, lane), c2);
    c3 = WMMA_F16(ap, load_B_nmajor(Vs, 48, 32, 0, lane), c3);

    __syncthreads();  // all waves done with Ks/Vs before next chunk's copy
  }

  // Row sums are spread across the 16 lanes of each half: reduce, invert.
#pragma unroll
  for (int r = 0; r < 8; ++r) {
    float v = rs[r];
    v += __shfl_xor(v, 1, 32);
    v += __shfl_xor(v, 2, 32);
    v += __shfl_xor(v, 4, 32);
    v += __shfl_xor(v, 8, 32);
    rs[r] = 1.0f / (v + 1e-8f);
  }

#pragma unroll
  for (int r = 0; r < 8; ++r) {
    const size_t o = (rowb + (size_t)(q0 + r + 8 * half)) * DMODEL + colh;
    out[o + n]      = c0[r] * rs[r];
    out[o + 16 + n] = c1[r] * rs[r];
    out[o + 32 + n] = c2[r] * rs[r];
    out[o + 48 + n] = c3[r] * rs[r];
  }
}

// ---------------------------------------------------------------------------
extern "C" void kernel_launch(void* const* d_in, const int* in_sizes, int n_in,
                              void* d_out, int out_size, void* d_ws,
                              size_t ws_size, hipStream_t stream) {
  (void)in_sizes; (void)n_in; (void)out_size; (void)ws_size;
  const float* Q    = (const float*)d_in[0];
  const float* mask = (const float*)d_in[1];
  const float* Wq   = (const float*)d_in[2];
  const float* bq   = (const float*)d_in[3];
  const float* Wk   = (const float*)d_in[4];
  const float* bk   = (const float*)d_in[5];
  const float* Wv   = (const float*)d_in[6];
  const float* bv   = (const float*)d_in[7];
  float* out = (float*)d_out;

  // Workspace (f16): Qh 4M | Wq/Wk/Wv 1M each | qh 4M | kh 4M | vt 4M = 38 MB
  _Float16* Qh  = (_Float16*)d_ws;
  _Float16* Wqh = Qh  + (size_t)ROWS * DMODEL;
  _Float16* Wkh = Wqh + (size_t)DMODEL * DMODEL;
  _Float16* Wvh = Wkh + (size_t)DMODEL * DMODEL;
  _Float16* qh  = Wvh + (size_t)DMODEL * DMODEL;
  _Float16* kh  = qh  + (size_t)ROWS * DMODEL;
  _Float16* vt  = kh  + (size_t)ROWS * DMODEL;

  const int nQ = ROWS * DMODEL;      // 4,194,304
  const int nW = DMODEL * DMODEL;    // 1,048,576
  cvt_f32_f16<<<nQ / (256 * 8), 256, 0, stream>>>(Q, Qh, nQ);
  cvt_f32_f16<<<nW / (256 * 8), 256, 0, stream>>>(Wq, Wqh, nW);
  cvt_f32_f16<<<nW / (256 * 8), 256, 0, stream>>>(Wk, Wkh, nW);
  cvt_f32_f16<<<nW / (256 * 8), 256, 0, stream>>>(Wv, Wvh, nW);

  // 16384 wave-tiles (256 row-tiles x 64 col-tiles), 8 waves per block.
  mha_qkv_proj<<<2048, 256, 0, stream>>>(Qh, Wqh, bq, Wkh, bk, Wvh, bv,
                                         qh, kh, vt);
  // 512 blocks = b(2) x h(16) x 16 query groups; 8 query tiles per block.
  mha_attention<<<512, 256, 0, stream>>>(qh, kh, vt, mask, out);
}
